// TSF_8710193676516
// MI455X (gfx1250) — matrix operand order repaired and checked by reading, not compiled, not run on hardware
//
#include <hip/hip_runtime.h>
#include <hip/hip_bf16.h>
#include <math.h>

// CDNA5 / gfx1250, wave32. Fused FAGCN FALayer:
//   e = tanh( PReLU( (h[dst]*h[src]) @ W1 + b1 ) @ W2 + b2 )
//   z = segment_sum( h[src] * e, dst )
// Edge GEMM on v_wmma_f32_16x16x32_bf16. Operands pre-swizzled into WMMA
// fragment order in LDS (fragment = two 16B-aligned ds_load_b128 per lane,
// conflict-free banking). GEMM stream is software-pipelined (double-buffered
// fragments) so WMMAs overlap the next fragment's LDS loads. Persistent grid
// amortizes the W1 LDS stage-in across ~3 edge tiles per block.

typedef __attribute__((ext_vector_type(16))) __bf16 v16bf;
typedef __attribute__((ext_vector_type(8)))  float  v8f;

#define DIM 192
#define KTILES 6        // 192 / 32
#define NTILES 12       // 192 / 16
#define NFRAGS (KTILES * NTILES)
#define NWAVES 8
#define TILE_M 16
#define EPB (NWAVES * TILE_M)   // 128 edges per block-tile
#define PGRID 1024              // persistent grid size

// Fragment-order LDS sizes (dwords): fragment = 2 planes * 32 lanes * 4 dwords
#define B_DWORDS (NFRAGS * 256)   // 73728 B
#define A_DWORDS (KTILES * 256)   // 6144 B per wave

__device__ __forceinline__ unsigned int pack2bf(float lo, float hi) {
  // round-to-nearest-even f32 -> bf16, packed pair (low half = first element)
  unsigned int a = __float_as_uint(lo);
  unsigned int b = __float_as_uint(hi);
  a = (a + 0x7FFFu + ((a >> 16) & 1u)) >> 16;
  b = (b + 0x7FFFu + ((b >> 16) & 1u)) >> 16;
  return (a & 0xFFFFu) | (b << 16);
}

__global__ void __launch_bounds__(256)
fagcn_zero_kernel(float4* __restrict__ p, long long n4) {
  long long i = (long long)blockIdx.x * blockDim.x + threadIdx.x;
  long long stride = (long long)gridDim.x * blockDim.x;
  for (; i < n4; i += stride) p[i] = make_float4(0.f, 0.f, 0.f, 0.f);
}

__global__ void __launch_bounds__(256)
fagcn_edge_kernel(const float* __restrict__ h,
                  const int*   __restrict__ src,
                  const int*   __restrict__ dst,
                  const float* __restrict__ W1,
                  const float* __restrict__ b1,
                  const float* __restrict__ alpha_p,
                  const float* __restrict__ W2,
                  const float* __restrict__ b2_p,
                  float* __restrict__ z,
                  float* __restrict__ eOut,
                  int Ecnt, int nTileBlocks)
{
  __shared__ unsigned int sBf[B_DWORDS];           // W1, fragment order
  __shared__ unsigned int sA[NWAVES][A_DWORDS];    // h2 tiles, fragment order
  __shared__ float sB1[DIM];
  __shared__ float sW2[DIM];
  __shared__ float sE[NWAVES][TILE_M];

  const int tid = threadIdx.x;

  // ---- Stage W1 into B-fragment order (once per persistent block) ----
  // dword i -> v4 = i&3, lane = (i>>2)&31, plane = (i>>7)&1, frag f = i>>8
  // fragment (kt,nt), lane l, v = plane*4+v4 holds bf16 pair:
  //   n = nt*16 + (l&15); pairIdx = kt*16 + (l>>4)*8 + v; K = 2*pairIdx, +1
  for (int i = tid; i < B_DWORDS; i += 256) {
    int v4 = i & 3;
    int ln = (i >> 2) & 31;
    int pl = (i >> 7) & 1;
    int f  = i >> 8;
    int kt = f / NTILES, nt = f - kt * NTILES;
    int v  = pl * 4 + v4;
    int n  = nt * 16 + (ln & 15);
    int pairIdx = kt * 16 + (ln >> 4) * 8 + v;
    int k0 = 2 * pairIdx;
    sBf[i] = pack2bf(W1[k0 * DIM + n], W1[(k0 + 1) * DIM + n]);
  }
  if (tid < DIM) { sB1[tid] = b1[tid]; sW2[tid] = W2[tid]; }
  __syncthreads();

  const int wave = tid >> 5;
  const int lane = tid & 31;
  const int half = lane >> 4;      // 0: lanes 0-15, 1: lanes 16-31
  const int l16  = lane & 15;
  const float alpha = alpha_p[0];
  const float b2v   = b2_p[0];

  const float4* h4 = (const float4*)h;   // row stride = 48 float4
  const uint4*  aq = (const uint4*)&sA[wave][0];
  const uint4*  bq = (const uint4*)&sBf[0];

  union Frag { v16bf v; uint4 u4[2]; };

  // ---- Persistent loop over 128-edge tile-blocks ----
  for (int bt = blockIdx.x; bt < nTileBlocks; bt += gridDim.x) {
    const int ebase = bt * EPB + wave * TILE_M;

    // Stage A tile: h2 = h[dst]*h[src], bf16 pairs in fragment order.
    // pair (row m, pairIdx pp): kt = pp>>4; rem = pp&15;
    //   lane = m + 16*(rem>>3); v = rem&7
    for (int m = 0; m < TILE_M; ++m) {
      int e = ebase + m;
      bool valid = (e < Ecnt);
      int srow = valid ? src[e] : 0;
      int drow = valid ? dst[e] : 0;
      for (int c = lane; c < 48; c += 32) {         // 48 float4 chunks per row
        unsigned int u0 = 0, u1 = 0;
        if (valid) {
          float4 a = h4[(long long)srow * 48 + c];
          float4 b = h4[(long long)drow * 48 + c];
          u0 = pack2bf(a.x * b.x, a.y * b.y);       // pair 2c
          u1 = pack2bf(a.z * b.z, a.w * b.w);       // pair 2c+1
        }
        #pragma unroll
        for (int t = 0; t < 2; ++t) {
          int pp  = 2 * c + t;
          int kt  = pp >> 4;
          int rem = pp & 15;
          int ln  = m + ((rem >> 3) << 4);
          int v   = rem & 7;
          sA[wave][((kt * 2 + (v >> 2)) * 32 + ln) * 4 + (v & 3)] = (t ? u1 : u0);
        }
      }
    }
    asm volatile("s_wait_dscnt 0" ::: "memory");  // wave-local LDS RAW fence

    // GEMM: [16 x 192] @ [192 x 192], double-buffered fragment pipeline
    v8f acc[NTILES];
    const v8f vzero = {0.f, 0.f, 0.f, 0.f, 0.f, 0.f, 0.f, 0.f};
    #pragma unroll
    for (int nt = 0; nt < NTILES; ++nt) acc[nt] = vzero;

    Frag acur, bcur;
    acur.u4[0] = aq[0 * 32 + lane];
    acur.u4[1] = aq[1 * 32 + lane];
    bcur.u4[0] = bq[0 * 32 + lane];
    bcur.u4[1] = bq[1 * 32 + lane];

    #pragma unroll
    for (int fi = 0; fi < NFRAGS; ++fi) {
      int nt = fi % NTILES;
      Frag anext = acur, bnext = bcur;
      if (fi + 1 < NFRAGS) {
        bnext.u4[0] = bq[((fi + 1) * 2 + 0) * 32 + lane];
        bnext.u4[1] = bq[((fi + 1) * 2 + 1) * 32 + lane];
        if (nt == NTILES - 1) {
          int ktn = fi / NTILES + 1;
          anext.u4[0] = aq[(ktn * 2 + 0) * 32 + lane];
          anext.u4[1] = aq[(ktn * 2 + 1) * 32 + lane];
        }
      }
      acc[nt] = __builtin_amdgcn_wmma_f32_16x16x32_bf16(
          false, acur.v, false, bcur.v, (short)0, acc[nt], false, false);
      acur = anext;
      bcur = bnext;
    }

    // Epilogue: +b1, PReLU, dot with W2, tanh
    // C layout: acc[nt][r] = g1[M = r + 8*half][N = nt*16 + l16]
    float part[8];
    #pragma unroll
    for (int r = 0; r < 8; ++r) part[r] = 0.f;
    #pragma unroll
    for (int nt = 0; nt < NTILES; ++nt) {
      int n = nt * 16 + l16;
      float b1n = sB1[n], w2n = sW2[n];
      #pragma unroll
      for (int r = 0; r < 8; ++r) {
        float g = acc[nt][r] + b1n;
        g = (g >= 0.f) ? g : alpha * g;
        part[r] += g * w2n;
      }
    }
    // reduce over N across the 16 lanes of each half (xor masks stay in-group)
    #pragma unroll
    for (int mask = 1; mask <= 8; mask <<= 1) {
      #pragma unroll
      for (int r = 0; r < 8; ++r) part[r] += __shfl_xor(part[r], mask, 32);
    }
    if (l16 == 0) {   // lane 0 -> rows 0..7, lane 16 -> rows 8..15
      #pragma unroll
      for (int r = 0; r < 8; ++r) {
        int m = half * 8 + r;
        float ev = tanhf(part[r] + b2v);
        sE[wave][m] = ev;
        int e = ebase + m;
        if (e < Ecnt) eOut[e] = ev;
      }
    }
    asm volatile("s_wait_dscnt 0" ::: "memory");

    // Scatter: z[dst] += h[src] * e (h and z are L2-resident)
    for (int m = 0; m < TILE_M; ++m) {
      int e = ebase + m;
      if (e >= Ecnt) continue;          // uniform per wave
      float ev = sE[wave][m];
      int srow = src[e], drow = dst[e];
      const float* hs = h + (long long)srow * DIM;
      float*       zd = z + (long long)drow * DIM;
      #pragma unroll
      for (int c = lane; c < DIM; c += 32)
        atomicAdd(&zd[c], hs[c] * ev);
    }
  }
}

extern "C" void kernel_launch(void* const* d_in, const int* in_sizes, int n_in,
                              void* d_out, int out_size, void* d_ws, size_t ws_size,
                              hipStream_t stream) {
  const float* h     = (const float*)d_in[0];
  const int*   src   = (const int*)  d_in[1];
  const int*   dst   = (const int*)  d_in[2];
  const float* W1    = (const float*)d_in[3];
  const float* b1    = (const float*)d_in[4];
  const float* alpha = (const float*)d_in[5];
  const float* W2    = (const float*)d_in[6];
  const float* b2    = (const float*)d_in[7];

  const int E = in_sizes[1];                 // 400000
  const int N = in_sizes[0] / DIM;           // 50000

  float* z    = (float*)d_out;               // [N, D]
  float* eOut = z + (long long)N * DIM;      // [E]

  // d_out is poisoned by the harness; z is accumulated with atomics -> zero it.
  long long n4 = (long long)N * DIM / 4;
  fagcn_zero_kernel<<<2048, 256, 0, stream>>>((float4*)z, n4);

  const int nTileBlocks = (E + EPB - 1) / EPB;   // 3125
  const int grid = nTileBlocks < PGRID ? nTileBlocks : PGRID;
  fagcn_edge_kernel<<<grid, 256, 0, stream>>>(h, src, dst, W1, b1, alpha, W2, b2,
                                              z, eOut, E, nTileBlocks);
}
